// GroupLinear_43903155700156
// MI455X (gfx1250) — matrix operand order repaired and checked by reading, not compiled, not run on hardware
//
#include <hip/hip_runtime.h>
#include <hip/hip_bf16.h>
#include <math.h>
#include <stdint.h>

typedef __bf16 bf16_t;
typedef bf16_t bf16x16 __attribute__((ext_vector_type(16)));
typedef float  floatx8 __attribute__((ext_vector_type(8)));

union BF16Frag { bf16x16 v; unsigned u[8]; };

// ---- CDNA5 async global->LDS copy (ASYNCcnt) -------------------------------
// LDS destination = LDS base + offset in the VDST VGPR; low 32 bits of a flat LDS pointer are the
// wave-relative LDS byte offset (aperture lives in the high half).
__device__ inline void async_copy_b128(void* lds_ptr, const void* gptr) {
    const unsigned lds_off = (unsigned)(uintptr_t)lds_ptr;
    asm volatile("global_load_async_to_lds_b128 %0, %1, off"
                 :: "v"(lds_off), "v"(gptr) : "memory");
}
__device__ inline void wait_async0() {
    asm volatile("s_wait_asynccnt 0x0" ::: "memory");
}

// ---- WMMA fragment loaders (wave32, 16x16x32 bf16) -------------------------
// A: LDS tile row-major [16 rows][lda], K base 0..31.
// ISA 16-bit A 16x32 layout: lanes 0-15 -> M=lane, K={hf*8 + 2p, +1} (p<4),
// K={16 + hf*8 + 2(p-4), +1} (p>=4), hf = lane>>4.
__device__ inline bf16x16 load_a_frag(const bf16_t* As, int lda, int lane) {
    const int hf = (lane >> 4) & 1;
    const int m  = lane & 15;
    const bf16_t* row = As + m * lda;
    BF16Frag f;
#pragma unroll
    for (int p = 0; p < 8; ++p) {
        const int k = ((p & 4) << 2) + (hf << 3) + ((p & 3) << 1);
        f.u[p] = *(const unsigned*)(row + k);
    }
    return f.v;
}

// B: LDS tile stored transposed [16 cols][ldb] (n-major, K contiguous).
// Lane holds column n = lane&15, K = hf*8 + e (e<8), 16 + hf*8 + (e-8) (e>=8).
__device__ inline bf16x16 load_b_frag(const bf16_t* Bst, int ldb, int lane) {
    const int hf = (lane >> 4) & 1;
    const int n  = lane & 15;
    const bf16_t* row = Bst + n * ldb + (hf << 3);
    BF16Frag f;
#pragma unroll
    for (int p = 0; p < 8; ++p) f.u[p] = *(const unsigned*)(row + ((p & 4) << 2) + ((p & 3) << 1));
    return f.v;
}

// ---- f32 -> bf16 weight conversion ----------------------------------------
__global__ __launch_bounds__(256) void f32_to_bf16_kernel(const float* __restrict__ src,
                                                          bf16_t* __restrict__ dst, int n) {
    int i = (blockIdx.x * blockDim.x + threadIdx.x) * 4;
    const int stride = gridDim.x * blockDim.x * 4;
    for (; i + 3 < n; i += stride) {
        float4 v = *(const float4*)(src + i);
        dst[i + 0] = (bf16_t)v.x;
        dst[i + 1] = (bf16_t)v.y;
        dst[i + 2] = (bf16_t)v.z;
        dst[i + 3] = (bf16_t)v.w;
    }
}

// ---- Grouped GEMM with fused channel-shuffle gather ------------------------
// Per group g: out[:, g*N + :] = relu( A[:, perm[g*K + :]] @ W[g] + bias[g] )
// Block: 128 threads = 4 waves, M tile = 64 (16 rows per wave). BK = 32.
// Each thread's gathered column is fixed (= lane), so perm is read once/k-step.
template <int KDIM, int NDIM, typename TA>
__global__ __launch_bounds__(128) void grouped_gemm_kernel(
    const TA* __restrict__ A, int lda, const int* __restrict__ perm,
    const bf16_t* __restrict__ W, const float* __restrict__ bias,
    bf16_t* __restrict__ out, int ldo) {
    constexpr int GLD = 40, NT = NDIM / 16;
    __shared__ bf16_t As[64 * GLD];
    __shared__ bf16_t Bst[NDIM * GLD];

    const int g    = blockIdx.y;
    const int m0   = blockIdx.x * 64;
    const int tid  = threadIdx.x;
    const int lane = tid & 31;
    const int wave = tid >> 5;  // 0..3 -> M sub-tile

    constexpr int CPR = NDIM / 8;           // 8-wide chunks per B row
    const int bkk = tid / CPR;              // B k-row base
    const int bnn = (tid % CPR) * 8;        // B n base

    floatx8 acc[NT] = {};

    for (int k0 = 0; k0 < KDIM; k0 += 32) {
        __syncthreads();
        // A fill: column c = lane is fixed -> one perm lookup per k-step
        const int src = perm[g * KDIM + k0 + lane];
        const TA* acol = A + (size_t)m0 * lda + src;
#pragma unroll
        for (int i = 0; i < 16; ++i) {
            const int r = wave + i * 4;
            As[r * GLD + lane] = (bf16_t)acol[(size_t)r * lda];
        }
        // B fill, transposed into LDS: read 8 contiguous n, scatter along k
#pragma unroll
        for (int i = 0; i < (32 * CPR) / 128; ++i) {
            const int kk = bkk + i * (128 / CPR);
            union { float4 v; bf16_t h[8]; } u;
            u.v = *(const float4*)(W + (size_t)g * KDIM * NDIM +
                                   (size_t)(k0 + kk) * NDIM + bnn);
#pragma unroll
            for (int j = 0; j < 8; ++j) Bst[(bnn + j) * GLD + kk] = u.h[j];
        }
        __syncthreads();

        const bf16x16 af = load_a_frag(As + wave * 16 * GLD, GLD, lane);
#pragma unroll
        for (int ni = 0; ni < NT; ++ni) {
            const bf16x16 bfg = load_b_frag(Bst + ni * 16 * GLD, GLD, lane);
            acc[ni] = __builtin_amdgcn_wmma_f32_16x16x32_bf16(
                false, af, false, bfg, (short)0, acc[ni], false, false);
        }
    }

    const int hf = (lane >> 4) & 1, nl = lane & 15;
#pragma unroll
    for (int ni = 0; ni < NT; ++ni)
#pragma unroll
        for (int r = 0; r < 8; ++r) {
            const int row = m0 + wave * 16 + hf * 8 + r;
            const int col = ni * 16 + nl;
            float v = acc[ni][r] + bias[g * NDIM + col];
            v = v > 0.f ? v : 0.f;  // ReLU
            out[(size_t)row * ldo + g * NDIM + col] = (bf16_t)v;
        }
}

// ---- Dense bf16 GEMM: C = A @ B + bias -------------------------------------
// 128x128 tile, BK=32, 256 threads = 8 waves (4 M x 2 N), 2x4 WMMA per wave.
// Double-buffered LDS: A tiles via global_load_async_to_lds_b128 (ASYNCcnt),
// B tiles prefetched into VGPRs and transpose-stored after compute.
__global__ __launch_bounds__(256) void gemm_bf16_kernel(
    const bf16_t* __restrict__ A, const bf16_t* __restrict__ B,
    const float* __restrict__ bias, bf16_t* __restrict__ C,
    int M, int N, int K) {
    constexpr int LD = 40;
    __shared__ bf16_t As[2][128 * LD];
    __shared__ bf16_t Bst[2][128 * LD];

    const int tid  = threadIdx.x;
    const int lane = tid & 31;
    const int wave = tid >> 5;
    const int wm   = wave & 3;   // M sub-tile (32 rows)
    const int wn   = wave >> 2;  // N sub-tile (64 cols)
    const int m0   = blockIdx.y * 128;
    const int n0   = blockIdx.x * 128;

    const int ar = tid >> 2;            // A fill row base (0..63, +64)
    const int ac = (tid & 3) << 3;      // A fill col (0,8,16,24)
    const int bk = tid >> 4;            // B fill k base (0..15, +16)
    const int bn = (tid & 15) << 3;     // B fill n base

    auto issueA = [&](int buf, int k0) {
#pragma unroll
        for (int i = 0; i < 2; ++i) {
            const int r = ar + i * 64;
            async_copy_b128(&As[buf][r * LD + ac],
                            A + (size_t)(m0 + r) * K + k0 + ac);
        }
    };
    auto loadB = [&](int k0, float4* regs) {
#pragma unroll
        for (int i = 0; i < 2; ++i)
            regs[i] = *(const float4*)(B + (size_t)(k0 + bk + i * 16) * N + n0 + bn);
    };
    auto storeB = [&](int buf, const float4* regs) {
#pragma unroll
        for (int i = 0; i < 2; ++i) {
            union { float4 v; bf16_t h[8]; } u;
            u.v = regs[i];
#pragma unroll
            for (int j = 0; j < 8; ++j)
                Bst[buf][(bn + j) * LD + bk + i * 16] = u.h[j];
        }
    };

    floatx8 acc[2][4] = {};
    float4 bregs[2];

    // Prologue: stage tile 0
    issueA(0, 0);
    loadB(0, bregs);
    storeB(0, bregs);
    wait_async0();
    __syncthreads();

    const int nk = K / 32;
    for (int it = 0; it < nk; ++it) {
        const int buf = it & 1;
        const bool has_next = (it + 1) < nk;
        if (has_next) {
            issueA(buf ^ 1, (it + 1) * 32);   // async A for next tile
            loadB((it + 1) * 32, bregs);      // B for next tile -> VGPRs
        }

        // Compute on current buffer
        bf16x16 af[2], bfg[4];
#pragma unroll
        for (int mi = 0; mi < 2; ++mi)
            af[mi] = load_a_frag(&As[buf][(wm * 32 + mi * 16) * LD], LD, lane);
#pragma unroll
        for (int ni = 0; ni < 4; ++ni)
            bfg[ni] = load_b_frag(&Bst[buf][(wn * 64 + ni * 16) * LD], LD, lane);
#pragma unroll
        for (int mi = 0; mi < 2; ++mi)
#pragma unroll
            for (int ni = 0; ni < 4; ++ni)
                acc[mi][ni] = __builtin_amdgcn_wmma_f32_16x16x32_bf16(
                    false, af[mi], false, bfg[ni], (short)0, acc[mi][ni], false, false);

        if (has_next) {
            storeB(buf ^ 1, bregs);  // transpose-store next B (buf^1 unread now)
            wait_async0();           // next A tile resident in LDS
            __syncthreads();         // all waves: next tile ready, current reads done
        }
    }

    const int hf = (lane >> 4) & 1, nl = lane & 15;
#pragma unroll
    for (int mi = 0; mi < 2; ++mi)
#pragma unroll
        for (int ni = 0; ni < 4; ++ni)
#pragma unroll
            for (int r = 0; r < 8; ++r) {
                const int row = m0 + wm * 32 + mi * 16 + hf * 8 + r;
                const int col = n0 + wn * 64 + ni * 16 + nl;
                const float v = acc[mi][ni][r] + bias[col];
                C[(size_t)row * N + col] = (bf16_t)v;
            }
}

// ---- Head: out = sigmoid(h4 @ W5 + b5), one wave per row -------------------
__global__ __launch_bounds__(256) void head_kernel(
    const bf16_t* __restrict__ h4, const float* __restrict__ W5,
    const float* __restrict__ b5, float* __restrict__ out, int Brows, int K) {
    const int row  = blockIdx.x * 8 + (threadIdx.x >> 5);
    const int lane = threadIdx.x & 31;
    if (row >= Brows) return;
    float s = 0.f;
    for (int k = lane; k < K; k += 32)
        s += (float)h4[(size_t)row * K + k] * W5[k];
#pragma unroll
    for (int off = 16; off > 0; off >>= 1) s += __shfl_xor(s, off, 32);
    if (lane == 0) out[row] = 1.f / (1.f + __expf(-(s + b5[0])));
}

// ---------------------------------------------------------------------------
extern "C" void kernel_launch(void* const* d_in, const int* in_sizes, int n_in,
                              void* d_out, int out_size, void* d_ws, size_t ws_size,
                              hipStream_t stream) {
    constexpr int B = 4096, G = 128;
    const float* x     = (const float*)d_in[0];
    const int*   perm1 = (const int*)  d_in[1];
    const int*   perm2 = (const int*)  d_in[2];
    const float* W1 = (const float*)d_in[3];
    const float* b1 = (const float*)d_in[4];
    const float* W2 = (const float*)d_in[5];
    const float* b2 = (const float*)d_in[6];
    const float* W3 = (const float*)d_in[7];
    const float* b3 = (const float*)d_in[8];
    const float* W4 = (const float*)d_in[9];
    const float* b4 = (const float*)d_in[10];
    const float* W5 = (const float*)d_in[11];
    const float* b5 = (const float*)d_in[12];

    char* ws = (char*)d_ws;
    size_t off = 0;
    auto carve = [&](size_t bytes) -> void* {
        void* p = ws + off;
        off += (bytes + 255) & ~(size_t)255;
        return p;
    };
    bf16_t* W1b = (bf16_t*)carve((size_t)G * 128 * 64 * 2);
    bf16_t* W2b = (bf16_t*)carve((size_t)G * 64 * 32 * 2);
    bf16_t* W3b = (bf16_t*)carve((size_t)4096 * 2048 * 2);
    bf16_t* W4b = (bf16_t*)carve((size_t)2048 * 1024 * 2);
    bf16_t* h1  = (bf16_t*)carve((size_t)B * 8192 * 2);
    bf16_t* h2  = (bf16_t*)carve((size_t)B * 4096 * 2);
    bf16_t* h3  = (bf16_t*)carve((size_t)B * 2048 * 2);
    bf16_t* h4  = (bf16_t*)carve((size_t)B * 1024 * 2);

    auto cvt = [&](const float* s, bf16_t* d, int n) {
        const int blocks = (n / 4 + 255) / 256;
        f32_to_bf16_kernel<<<blocks, 256, 0, stream>>>(s, d, n);
    };
    cvt(W1, W1b, G * 128 * 64);
    cvt(W2, W2b, G * 64 * 32);
    cvt(W3, W3b, 4096 * 2048);
    cvt(W4, W4b, 2048 * 1024);

    // Stage 1: shuffle(perm1) + grouped [4096x128]@[128x64] + ReLU -> h1 (bf16)
    grouped_gemm_kernel<128, 64, float>
        <<<dim3(B / 64, G), 128, 0, stream>>>(x, G * 128, perm1, W1b, b1, h1, G * 64);
    // Stage 2: shuffle(perm2) + grouped [4096x64]@[64x32] + ReLU -> h2 (bf16)
    grouped_gemm_kernel<64, 32, bf16_t>
        <<<dim3(B / 64, G), 128, 0, stream>>>(h1, G * 64, perm2, W2b, b2, h2, G * 32);
    // Stage 3: [4096,4096]@[4096,2048] + b3 -> h3
    gemm_bf16_kernel<<<dim3(2048 / 128, B / 128), 256, 0, stream>>>(
        h2, W3b, b3, h3, B, 2048, 4096);
    // Stage 4: [4096,2048]@[2048,1024] + b4 -> h4
    gemm_bf16_kernel<<<dim3(1024 / 128, B / 128), 256, 0, stream>>>(
        h3, W4b, b4, h4, B, 1024, 2048);
    // Stage 5: sigmoid(h4 @ W5 + b5) -> d_out [4096]
    head_kernel<<<B / 8, 256, 0, stream>>>(h4, W5, b5, (float*)d_out, B, 1024);
}
